// GRU_77266461655247
// MI455X (gfx1250) — compile-verified, matrix operand order checked by reading, not run on hardware
//
#include <hip/hip_runtime.h>
#include <hip/hip_bf16.h>

// ---------------------------------------------------------------------------
// 2-layer GRU, B=32, T=2048, I=128, H=256  (PyTorch gate order r,z,n)
//
// Plan (MI455X / gfx1250, wave32, WMMA):
//  1) pack weights + x into WMMA bf16 operand layouts (elementwise kernels)
//  2) xW = X @ Wih^T  via v_wmma_f32_16x16x32_bf16 (full grid), stored bf16
//     in C-tile layout (~100MB -> resident in 192MB L2 for the recurrence)
//  3) recurrence: ONE 512-thread workgroup; Whh held in VGPRs (24 v16bf
//     tiles/wave), h kept in LDS both as f32 and as A-operand bf16 tiles;
//     per step: 48 WMMAs/wave + f32 gate nonlinearity + 2 barriers.
//  layer0 recurrence emits y0 already A-packed for layer1's xW GEMM;
//  layer1 recurrence writes y1 directly to d_out.
// ---------------------------------------------------------------------------

typedef __attribute__((ext_vector_type(16))) __bf16 v16bf;
typedef __attribute__((ext_vector_type(8)))  __bf16 v8bf;
typedef __attribute__((ext_vector_type(8)))  float  v8f;

#define GB   32      // batch
#define GT   2048    // timesteps
#define GI   128     // input size (layer 0)
#define GH   256     // hidden
#define GNG  768     // 3*H
#define GNT  48      // N tiles (768/16)
#define GMT  4096    // (B*T)/16 global M tiles

__device__ __forceinline__ float sigmoidf_(float x) {
    return 1.0f / (1.0f + __expf(-x));
}
__device__ __forceinline__ float tanhf_(float x) {
    return 1.0f - 2.0f / (__expf(2.0f * x) + 1.0f);
}

// --- WMMA bf16 operand slot math (ISA 7.12.2, 16-bit A 16x32 / B 32x16) ----
// A element (m,k): lanes 0-15 hold K {0..7,16..23}, lanes 16-31 {8..15,24..31}
// B element (k,n): mirrored with n -> lane low bits. Tile = 512 bf16,
// per-lane 16 contiguous bf16 (so operand load = 2 x b128).

// ---------------------------------------------------------------------------
// Pack a [3H x K] f32 weight into B-operand bf16 tiles: tile (nt,kt) of 512.
__global__ void pack_w_kernel(const float* __restrict__ W,
                              __bf16* __restrict__ P, int K, int Kt) {
    int idx = blockIdx.x * blockDim.x + threadIdx.x;
    if (idx >= GNG * K) return;
    int n = idx / K, k = idx % K;
    int kt = k >> 5, kk = k & 31;
    int hi = (kk >> 3) & 1;
    int lane = (n & 15) + (hi << 4);
    int el = (kk & 7) + ((kk >= 16) ? 8 : 0);
    P[(size_t)((n >> 4) * Kt + kt) * 512 + lane * 16 + el] = (__bf16)W[idx];
}

// Pack x [B,T,I] f32 into A-operand bf16 tiles with row m = t*32+b.
__global__ void pack_x_kernel(const float* __restrict__ x,
                              __bf16* __restrict__ P) {
    int idx = blockIdx.x * blockDim.x + threadIdx.x;
    if (idx >= GB * GT * GI) return;
    int b = idx / (GT * GI);
    int r = idx % (GT * GI);
    int t = r / GI, i = r % GI;
    int m = t * GB + b, k = i;
    int kt = k >> 5, kk = k & 31;
    int hi = (kk >> 3) & 1;
    int lane = (m & 15) + (hi << 4);
    int el = (kk & 7) + ((kk >= 16) ? 8 : 0);
    P[(size_t)((m >> 4) * 4 + kt) * 512 + lane * 16 + el] = (__bf16)x[idx];
}

// ---------------------------------------------------------------------------
// C[MtxNT tiles] = A @ B  (bf16 WMMA, f32 accum, bf16 C-tile output).
// One 16x16 output tile per wave; operands pre-packed -> pure b128 + wmma.
__global__ __launch_bounds__(256)
void wmma_gemm(const __bf16* __restrict__ Ap, const __bf16* __restrict__ Bp,
               __bf16* __restrict__ Cp, int Kt, int MtTot) {
    int w = (blockIdx.x * blockDim.x + threadIdx.x) >> 5;
    int lane = threadIdx.x & 31;
    int Mt = w / GNT, nt = w % GNT;
    if (Mt >= MtTot) return;                   // wave-uniform
    const v16bf* At = (const v16bf*)(Ap + (size_t)Mt * Kt * 512);
    const v16bf* Bt = (const v16bf*)(Bp + (size_t)nt * Kt * 512);
    v8f acc = {};
    for (int kt = 0; kt < Kt; ++kt) {
        acc = __builtin_amdgcn_wmma_f32_16x16x32_bf16(
            false, At[kt * 32 + lane], false, Bt[kt * 32 + lane],
            (short)0, acc, false, false);
    }
    v8bf o;
#pragma unroll
    for (int i = 0; i < 8; ++i) o[i] = (__bf16)acc[i];
    *((v8bf*)(Cp + (size_t)(Mt * GNT + nt) * 256) + lane) = o;
}

// ---------------------------------------------------------------------------
// Persistent single-workgroup GRU recurrence for one layer.
// 512 threads = 16 waves; wave jt owns gate columns [jt*16, jt*16+16) for
// r,z,n and both 16-row batch halves. Whh tiles live in VGPRs for all T.
#define HSTR 264   // padded f32 row stride in LDS (bank-conflict avoidance)

__global__ __launch_bounds__(512)
void gru_recur(const __bf16* __restrict__ xW,     // C-tile bf16 [T*2 x 48]
               const __bf16* __restrict__ WhhP,   // B-packed bf16 [48 x 8]
               const float* __restrict__ h0,      // [32,256] f32
               const float* __restrict__ bih,
               const float* __restrict__ bhh,
               float* __restrict__ yOut,          // layer1: y1 [B,T,H] else 0
               __bf16* __restrict__ yPack,        // layer0: A-packed y else 0
               float* __restrict__ hOut) {        // [32,256] final hidden
    __shared__ __align__(16) __bf16 sA[2 * 8 * 512];  // h, A-operand layout
    __shared__ float sH[32 * HSTR];                   // h, f32

    const int tid = threadIdx.x;
    const int jt = tid >> 5;       // 0..15
    const int lane = tid & 31;

    // ---- persistent B tiles (r,z,n) x 8 K-tiles: 192 VGPRs/wave ----
    v16bf Bt[3][8];
#pragma unroll
    for (int g = 0; g < 3; ++g) {
        const int nt = g * 16 + jt;
#pragma unroll
        for (int kt = 0; kt < 8; ++kt)
            Bt[g][kt] = *((const v16bf*)(WhhP + (size_t)(nt * 8 + kt) * 512) + lane);
    }

    const int colj = jt * 16 + (lane & 15);
    const float bR  = bih[colj] + bhh[colj];
    const float bZ  = bih[256 + colj] + bhh[256 + colj];
    const float bNi = bih[512 + colj];
    const float bNh = bhh[512 + colj];

    // ---- init hidden state in LDS (f32 + A-packed bf16) ----
    for (int idx = tid; idx < 32 * 256; idx += 512) {
        const int m = idx >> 8, k = idx & 255;
        const float v = h0[idx];
        sH[m * HSTR + k] = v;
        const int kt = k >> 5, kk = k & 31;
        const int hi = (kk >> 3) & 1;
        const int la = (m & 15) + (hi << 4);
        const int el = (kk & 7) + ((kk >= 16) ? 8 : 0);
        sA[((m >> 4) * 8 + kt) * 512 + la * 16 + el] = (__bf16)v;
    }
    __syncthreads();

    // A-pack slot constants for our output column colj
    const int kc = colj >> 5;
    const int kkc = colj & 31;
    const int hic = (kkc >> 3) & 1;
    const int elc = (kkc & 7) + ((kkc >= 16) ? 8 : 0);

    const v8f vz = {};

    for (int t = 0; t < GT; ++t) {
        const size_t Mt0 = (size_t)t * 2;

        // prefetch this step's input projections (bf16, C layout, L2-hot)
        v8bf xw[2][3];
#pragma unroll
        for (int ml = 0; ml < 2; ++ml)
#pragma unroll
            for (int g = 0; g < 3; ++g)
                xw[ml][g] =
                    *((const v8bf*)(xW + ((Mt0 + ml) * GNT + g * 16 + jt) * 256) + lane);

        v8f acc[2][3];
#pragma unroll
        for (int ml = 0; ml < 2; ++ml)
#pragma unroll
            for (int g = 0; g < 3; ++g) acc[ml][g] = vz;

        // gh = h @ Whh^T : 48 WMMAs/wave, A from LDS, B from VGPRs
#pragma unroll
        for (int kt = 0; kt < 8; ++kt) {
            const v16bf a0 = *((const v16bf*)(sA + (0 * 8 + kt) * 512) + lane);
            const v16bf a1 = *((const v16bf*)(sA + (1 * 8 + kt) * 512) + lane);
#pragma unroll
            for (int g = 0; g < 3; ++g) {
                acc[0][g] = __builtin_amdgcn_wmma_f32_16x16x32_bf16(
                    false, a0, false, Bt[g][kt], (short)0, acc[0][g], false, false);
                acc[1][g] = __builtin_amdgcn_wmma_f32_16x16x32_bf16(
                    false, a1, false, Bt[g][kt], (short)0, acc[1][g], false, false);
            }
        }

        // old h (f32) for z*h term
        float ho[2][8];
#pragma unroll
        for (int ml = 0; ml < 2; ++ml) {
            const int row0 = ml * 16 + ((lane >> 4) << 3);
#pragma unroll
            for (int v = 0; v < 8; ++v) ho[ml][v] = sH[(row0 + v) * HSTR + colj];
        }
        __syncthreads();  // all reads of the old state done

        // gates + state update; write back f32 + A-packed bf16 (+ layer out)
#pragma unroll
        for (int ml = 0; ml < 2; ++ml) {
            const int row0 = ml * 16 + ((lane >> 4) << 3);
#pragma unroll
            for (int v = 0; v < 8; ++v) {
                const float r = sigmoidf_((float)xw[ml][0][v] + bR + acc[ml][0][v]);
                const float z = sigmoidf_((float)xw[ml][1][v] + bZ + acc[ml][1][v]);
                const float n =
                    tanhf_((float)xw[ml][2][v] + bNi + r * (acc[ml][2][v] + bNh));
                const float hn = (1.0f - z) * n + z * ho[ml][v];
                const int m = row0 + v;  // batch row 0..31
                sH[m * HSTR + colj] = hn;
                const int la = (m & 15) + (hic << 4);
                sA[(ml * 8 + kc) * 512 + la * 16 + elc] = (__bf16)hn;
                if (yPack)  // layer0 -> A-packed input for layer1 GEMM
                    yPack[((Mt0 + ml) * 8 + kc) * 512 + la * 16 + elc] = (__bf16)hn;
                if (yOut)   // layer1 -> y1[b][t][j]
                    yOut[((size_t)m * GT + t) * GH + colj] = hn;
            }
        }
        __syncthreads();  // new state visible before next step's A loads
    }

    for (int idx = tid; idx < 32 * 256; idx += 512)
        hOut[idx] = sH[(idx >> 8) * HSTR + (idx & 255)];
}

// ---------------------------------------------------------------------------
extern "C" void kernel_launch(void* const* d_in, const int* in_sizes, int n_in,
                              void* d_out, int out_size, void* d_ws,
                              size_t ws_size, hipStream_t stream) {
    (void)in_sizes; (void)n_in; (void)out_size; (void)ws_size;

    const float* x    = (const float*)d_in[0];   // [32,2048,128]
    const float* h_0  = (const float*)d_in[1];   // [2,32,256]
    const float* Wih0 = (const float*)d_in[2];   // [768,128]
    const float* Whh0 = (const float*)d_in[3];   // [768,256]
    const float* bih0 = (const float*)d_in[4];
    const float* bhh0 = (const float*)d_in[5];
    const float* Wih1 = (const float*)d_in[6];   // [768,256]
    const float* Whh1 = (const float*)d_in[7];
    const float* bih1 = (const float*)d_in[8];
    const float* bhh1 = (const float*)d_in[9];

    float* y1    = (float*)d_out;                        // [32,2048,256]
    float* hOut0 = y1 + (size_t)GB * GT * GH;            // h_n[0]
    float* hOut1 = hOut0 + (size_t)GB * GH;              // h_n[1]

    // workspace layout (bytes)
    char* ws = (char*)d_ws;
    __bf16* WihP0 = (__bf16*)(ws);                        //   192 KB
    __bf16* WhhP0 = (__bf16*)(ws + 196608);               //   384 KB
    __bf16* WihP1 = (__bf16*)(ws + 589824);               //   384 KB
    __bf16* WhhP1 = (__bf16*)(ws + 983040);               //   384 KB
    __bf16* Xp0   = (__bf16*)(ws + 1376256);              //  16 MB
    __bf16* Xp1   = (__bf16*)(ws + 18153472);             //  32 MB
    __bf16* xWp   = (__bf16*)(ws + 51707904);             // 100 MB (reused)
    // total ~152.4 MB

    // 1) pack weights into WMMA B-operand layout
    pack_w_kernel<<<(GNG * GI + 255) / 256, 256, 0, stream>>>(Wih0, WihP0, GI, 4);
    pack_w_kernel<<<(GNG * GH + 255) / 256, 256, 0, stream>>>(Whh0, WhhP0, GH, 8);
    pack_w_kernel<<<(GNG * GH + 255) / 256, 256, 0, stream>>>(Wih1, WihP1, GH, 8);
    pack_w_kernel<<<(GNG * GH + 255) / 256, 256, 0, stream>>>(Whh1, WhhP1, GH, 8);

    // 2) pack x into A-operand layout (rows m = t*32+b)
    pack_x_kernel<<<(GB * GT * GI + 255) / 256, 256, 0, stream>>>(x, Xp0);

    // 3) layer 0: xW0 = X @ Wih0^T  (4096*48 tiles, 8 waves/block)
    wmma_gemm<<<(GMT * GNT) / 8, 256, 0, stream>>>(Xp0, WihP0, xWp, 4, GMT);

    // 4) layer 0 recurrence (emits y0 A-packed into Xp1)
    gru_recur<<<1, 512, 0, stream>>>(xWp, WhhP0, h_0, bih0, bhh0,
                                     nullptr, Xp1, hOut0);

    // 5) layer 1: xW1 = Y0 @ Wih1^T  (reuses xWp)
    wmma_gemm<<<(GMT * GNT) / 8, 256, 0, stream>>>(Xp1, WihP1, xWp, 8, GMT);

    // 6) layer 1 recurrence (writes y1 straight to d_out)
    gru_recur<<<1, 512, 0, stream>>>(xWp, WhhP1, h_0 + (size_t)GB * GH,
                                     bih1, bhh1, y1, nullptr, hOut1);
}
